// GATConvLayer_5592047419416
// MI455X (gfx1250) — matrix operand order repaired and checked by reading, not compiled, not run on hardware
//
#include <hip/hip_runtime.h>
#include <hip/hip_bf16.h>

typedef __attribute__((ext_vector_type(16))) _Float16 v16h;
typedef __attribute__((ext_vector_type(8)))  _Float16 v8h;
typedef __attribute__((ext_vector_type(4)))  _Float16 v4h;
typedef __attribute__((ext_vector_type(8)))  float    v8f;
typedef __attribute__((ext_vector_type(4)))  unsigned int u32x4;
typedef __attribute__((ext_vector_type(8)))  int      i32x8;
typedef __attribute__((ext_vector_type(4)))  int      i32x4;

#define GAT_IN_CH   256
#define GAT_OUT_TOT 128   // HEADS * OUT_CH
#define GAT_HEADS   4
#define GAT_OUT_CH  32
#define GAT_EPS     1e-8f
#define GAT_SLOPE   0.2f

#if defined(__HIP_DEVICE_COMPILE__) && __has_builtin(__builtin_amdgcn_tensor_load_to_lds)
#define GAT_HAVE_TDM 1
#else
#define GAT_HAVE_TDM 0
#endif

// ---------------------------------------------------------------------------
// K0: init accumulators (out = 0, alpha_max = -inf, alpha_sum = 0)
// ---------------------------------------------------------------------------
__global__ void gat_init(float* __restrict__ out, float* __restrict__ a_max,
                         float* __restrict__ a_sum, int nNodes) {
    int i = blockIdx.x * blockDim.x + threadIdx.x;
    if (i < nNodes * GAT_OUT_CH) out[i] = 0.0f;
    if (i < nNodes * GAT_HEADS) {
        a_max[i] = -__builtin_inff();
        a_sum[i] = 0.0f;
    }
}

// ---------------------------------------------------------------------------
// Kc: one-pass fp32 -> fp16 convert of x and W (xh stays L2-resident; halves
// all downstream gather traffic and lets GEMM A-frags be 2x b128 loads).
// ---------------------------------------------------------------------------
__global__ __launch_bounds__(256) void gat_convert(
    const float* __restrict__ x, const float* __restrict__ W,
    _Float16* __restrict__ xh, _Float16* __restrict__ Wh, int nx4, int nw4)
{
    int i = blockIdx.x * 256 + threadIdx.x;
    if (i < nx4) {
        float4 v = *(const float4*)(x + (size_t)i * 4);
        v4h o = { (_Float16)v.x, (_Float16)v.y, (_Float16)v.z, (_Float16)v.w };
        *(v4h*)(xh + (size_t)i * 4) = o;
    }
    if (i < nw4) {
        float4 v = *(const float4*)(W + (size_t)i * 4);
        v4h o = { (_Float16)v.x, (_Float16)v.y, (_Float16)v.z, (_Float16)v.w };
        *(v4h*)(Wh + (size_t)i * 4) = o;
    }
}

// ---------------------------------------------------------------------------
// K1: h16[N,128] = x[N,256] @ W^T  via v_wmma_f32_16x16x32_f16.
// 256 threads = 8 waves; wave handles 16 rows x 128 cols; K-loop of 8.
// W (f16, 128x256 = 64 KB) staged into LDS by the Tensor Data Mover.
// ---------------------------------------------------------------------------
__global__ __launch_bounds__(256) void gat_gemm_h(
    const _Float16* __restrict__ xh, const _Float16* __restrict__ Wh,
    _Float16* __restrict__ h16, int nNodes)
{
    __shared__ _Float16 Wl[GAT_OUT_TOT * GAT_IN_CH]; // 64 KB, linear [n][k]

#if GAT_HAVE_TDM
    if (threadIdx.x < 32) {   // one wave issues the DMA; EXEC is ignored by TDM
        unsigned lds = (unsigned)(size_t)(&Wl[0]);   // low 32 bits = LDS offset
        unsigned long long ga = (unsigned long long)(size_t)Wh;
        // D# group 0: count=1 | lds_addr | global_addr[56:0] | type=2
        u32x4 g0 = { 1u, lds, (unsigned)(ga & 0xFFFFFFFFu),
                     (unsigned)((ga >> 32) & 0x01FFFFFFu) | (2u << 30) };
        // D# group 1: data_size=2B; tensor 256x128; tile 256x128; stride 256
        i32x8 g1 = { (int)(1u << 16),          // data_size=1 (2 bytes)
                     (int)(256u << 16),        // tensor_dim0[15:0] in [31:16]
                     (int)(128u << 16),        // dim0 hi=0 | tensor_dim1[15:0]
                     (int)(256u << 16),        // dim1 hi=0 | tile_dim0=256
                     (int)128,                 // tile_dim1=128 | tile_dim2=0
                     (int)256,                 // tensor_dim0_stride[31:0]
                     0, 0 };                   // stride hi | tensor_dim1_stride
        i32x4 z4 = { 0, 0, 0, 0 };
#if defined(__clang_major__) && (__clang_major__ >= 23)
        i32x8 z8 = { 0, 0, 0, 0, 0, 0, 0, 0 };
        __builtin_amdgcn_tensor_load_to_lds(g0, g1, z4, z4, z8, 0);
#else
        __builtin_amdgcn_tensor_load_to_lds(g0, g1, z4, z4, 0);
#endif
        __builtin_amdgcn_s_wait_tensorcnt(0);
    }
#else
    for (int i = threadIdx.x; i < GAT_OUT_TOT * GAT_IN_CH; i += 256)
        Wl[i] = Wh[i];
#endif
    __syncthreads();

    const int lane    = threadIdx.x & 31;
    const int wave    = threadIdx.x >> 5;
    const int rowBase = blockIdx.x * 128 + wave * 16;
    const int hiHalf  = lane >> 4;       // 0 or 1 (lane group)
    const int nIdx    = lane & 15;

    v8f acc[8] = {};

    int arow = rowBase + nIdx;
    if (arow >= nNodes) arow = nNodes - 1;   // clamp; stores are guarded
    const _Float16* __restrict__ xrow = xh + (size_t)arow * GAT_IN_CH;

    for (int kk = 0; kk < GAT_IN_CH; kk += 32) {
        // A fragment 16x32 f16: lane<16 holds K {kk..kk+7, kk+16..kk+23},
        // lane>=16 holds K {kk+8..kk+15, kk+24..kk+31} -> two b128 loads.
        v8h alo = *(const v8h*)(xrow + kk + (hiHalf << 3));
        v8h ahi = *(const v8h*)(xrow + kk + 16 + (hiHalf << 3));
        v16h a = __builtin_shufflevector(alo, ahi,
                 0, 1, 2, 3, 4, 5, 6, 7, 8, 9, 10, 11, 12, 13, 14, 15);

        // B fragments: lane holds col n = t*16 + nIdx, K {kb..kb+7, kb+16..kb+23}
        // with kb = kk + hiHalf*8 (matches A). Pipeline tile t+1's loads past wmma(t).
        const int kb = kk + (hiHalf << 3);
        const _Float16* bbase = &Wl[nIdx * GAT_IN_CH + kb];
        v8h blo = *(const v8h*)(bbase);
        v8h bhi = *(const v8h*)(bbase + 16);
        #pragma unroll
        for (int t = 0; t < 8; ++t) {
            v8h nlo = blo, nhi = bhi;
            if (t < 7) {
                const _Float16* bn = bbase + (size_t)(t + 1) * 16 * GAT_IN_CH;
                nlo = *(const v8h*)(bn);
                nhi = *(const v8h*)(bn + 16);
            }
            v16h b = __builtin_shufflevector(blo, bhi,
                     0, 1, 2, 3, 4, 5, 6, 7, 8, 9, 10, 11, 12, 13, 14, 15);
            acc[t] = __builtin_amdgcn_wmma_f32_16x16x32_f16(
                         false, a, false, b, (short)0, acc[t], false, false);
            blo = nlo; bhi = nhi;
        }
    }

    // D layout: VGPR r -> row = rowBase + r + 8*hiHalf, col = t*16 + nIdx.
    if (rowBase + 16 <= nNodes) {            // uniform fast path, no guards
        #pragma unroll
        for (int t = 0; t < 8; ++t) {
            int col = t * 16 + nIdx;
            #pragma unroll
            for (int r = 0; r < 8; ++r) {
                int row = rowBase + r + (hiHalf << 3);
                h16[(size_t)row * GAT_OUT_TOT + col] = (_Float16)acc[t][r];
            }
        }
    } else {
        #pragma unroll
        for (int t = 0; t < 8; ++t) {
            int col = t * 16 + nIdx;
            #pragma unroll
            for (int r = 0; r < 8; ++r) {
                int row = rowBase + r + (hiHalf << 3);
                if (row < nNodes)
                    h16[(size_t)row * GAT_OUT_TOT + col] = (_Float16)acc[t][r];
            }
        }
    }
}

// ---------------------------------------------------------------------------
// K2: alpha_src[n,h] = dot(h[n,h*32:], att_src[h]); same for dst.
// One wave per node; lane holds 4 channels; 8-lane head-group reduction.
// ---------------------------------------------------------------------------
__global__ __launch_bounds__(256) void gat_alpha_node(
    const _Float16* __restrict__ h16, const float* __restrict__ att_src,
    const float* __restrict__ att_dst, float* __restrict__ a_src,
    float* __restrict__ a_dst, int nNodes)
{
    int node = (int)((blockIdx.x * 256 + threadIdx.x) >> 5);
    int lane = threadIdx.x & 31;
    if (node >= nNodes) return;

    v4h hv = *(const v4h*)(h16 + (size_t)node * GAT_OUT_TOT + lane * 4);
    float4 as = *(const float4*)(att_src + lane * 4);
    float4 ad = *(const float4*)(att_dst + lane * 4);
    float h0 = (float)hv[0], h1 = (float)hv[1], h2 = (float)hv[2], h3 = (float)hv[3];
    float ps = h0 * as.x + h1 * as.y + h2 * as.z + h3 * as.w;
    float pd = h0 * ad.x + h1 * ad.y + h2 * ad.z + h3 * ad.w;
    #pragma unroll
    for (int m = 1; m < 8; m <<= 1) {
        ps += __shfl_xor(ps, m, 32);
        pd += __shfl_xor(pd, m, 32);
    }
    if ((lane & 7) == 0) {
        int head = lane >> 3;
        a_src[node * GAT_HEADS + head] = ps;
        a_dst[node * GAT_HEADS + head] = pd;
    }
}

// ---------------------------------------------------------------------------
// K3: per-edge LeakyReLU logit + order-preserving float atomic max on col.
// ---------------------------------------------------------------------------
__global__ void gat_edge_alpha(
    const int* __restrict__ erow, const int* __restrict__ ecol,
    const float* __restrict__ a_src, const float* __restrict__ a_dst,
    float* __restrict__ alphaE, float* __restrict__ a_max, int nEdges)
{
    int e = blockIdx.x * blockDim.x + threadIdx.x;
    if (e >= nEdges) return;
    int r = erow[e], c = ecol[e];
    float4 s = *(const float4*)(a_src + (size_t)r * GAT_HEADS);
    float4 d = *(const float4*)(a_dst + (size_t)c * GAT_HEADS);
    float v[4] = { s.x + d.x, s.y + d.y, s.z + d.z, s.w + d.w };
    #pragma unroll
    for (int hh = 0; hh < GAT_HEADS; ++hh) {
        float a = v[hh];
        a = (a >= 0.0f) ? a : GAT_SLOPE * a;
        v[hh] = a;
        int* mp = (int*)(a_max + (size_t)c * GAT_HEADS + hh);
        if (a >= 0.0f) atomicMax(mp, __float_as_int(a));
        else           atomicMin((unsigned int*)mp, __float_as_uint(a));
    }
    *(float4*)(alphaE + (size_t)e * GAT_HEADS) = make_float4(v[0], v[1], v[2], v[3]);
}

// ---------------------------------------------------------------------------
// K4: alphaE = exp(alphaE - max[col]); atomicAdd into sum[col].
// ---------------------------------------------------------------------------
__global__ void gat_edge_exp(
    const int* __restrict__ ecol, const float* __restrict__ a_max,
    float* __restrict__ alphaE, float* __restrict__ a_sum, int nEdges)
{
    int e = blockIdx.x * blockDim.x + threadIdx.x;
    if (e >= nEdges) return;
    int c = ecol[e];
    float4 a = *(const float4*)(alphaE + (size_t)e * GAT_HEADS);
    float4 m = *(const float4*)(a_max + (size_t)c * GAT_HEADS);
    float e0 = __expf(a.x - m.x);
    float e1 = __expf(a.y - m.y);
    float e2 = __expf(a.z - m.z);
    float e3 = __expf(a.w - m.w);
    *(float4*)(alphaE + (size_t)e * GAT_HEADS) = make_float4(e0, e1, e2, e3);
    float* sp = a_sum + (size_t)c * GAT_HEADS;
    atomicAdd(sp + 0, e0);
    atomicAdd(sp + 1, e1);
    atomicAdd(sp + 2, e2);
    atomicAdd(sp + 3, e3);
}

// ---------------------------------------------------------------------------
// K5: out[col,:] += 0.25 * h16[row, head, :] * alpha  (head-mean folded in).
// One wave per edge; lane = 4 channels (8-byte f16 gather); head = lane>>3.
// h16 (25.6 MB) is L2-resident -> gathers served from L2, not HBM.
// ---------------------------------------------------------------------------
__global__ __launch_bounds__(256) void gat_scatter(
    const int* __restrict__ erow, const int* __restrict__ ecol,
    const _Float16* __restrict__ h16, const float* __restrict__ alphaE,
    const float* __restrict__ a_sum, float* __restrict__ out, int nEdges)
{
    int e = (int)((blockIdx.x * 256 + threadIdx.x) >> 5);
    int lane = threadIdx.x & 31;
    if (e >= nEdges) return;
    int r = erow[e], c = ecol[e];
    int head = lane >> 3;
    float w = alphaE[(size_t)e * GAT_HEADS + head] /
              (a_sum[(size_t)c * GAT_HEADS + head] + GAT_EPS) * 0.25f;
    v4h hv = *(const v4h*)(h16 + (size_t)r * GAT_OUT_TOT + lane * 4);
    float* op = out + (size_t)c * GAT_OUT_CH + (lane & 7) * 4;
    atomicAdd(op + 0, (float)hv[0] * w);
    atomicAdd(op + 1, (float)hv[1] * w);
    atomicAdd(op + 2, (float)hv[2] * w);
    atomicAdd(op + 3, (float)hv[3] * w);
}

// ---------------------------------------------------------------------------
extern "C" void kernel_launch(void* const* d_in, const int* in_sizes, int n_in,
                              void* d_out, int out_size, void* d_ws, size_t ws_size,
                              hipStream_t stream) {
    const float* x       = (const float*)d_in[0];
    const int*   ei      = (const int*)  d_in[1];
    const float* W       = (const float*)d_in[2];
    const float* att_src = (const float*)d_in[3];
    const float* att_dst = (const float*)d_in[4];
    float* out = (float*)d_out;

    const int nNodes = in_sizes[0] / GAT_IN_CH;
    const int nEdges = in_sizes[1] / 2;
    const int* erow = ei;
    const int* ecol = ei + nEdges;

    // workspace: f16 region first, then 16B-aligned fp32 region
    _Float16* hbase = (_Float16*)d_ws;
    size_t ho = 0;
    _Float16* h16 = hbase + ho; ho += (size_t)nNodes * GAT_OUT_TOT;
    _Float16* xh  = hbase + ho; ho += (size_t)nNodes * GAT_IN_CH;
    _Float16* Wh  = hbase + ho; ho += (size_t)GAT_OUT_TOT * GAT_IN_CH;
    ho = (ho + 7) & ~(size_t)7;
    float* fbase = (float*)(hbase + ho);
    size_t fo = 0;
    float* a_src  = fbase + fo; fo += (size_t)nNodes * GAT_HEADS;
    float* a_dst  = fbase + fo; fo += (size_t)nNodes * GAT_HEADS;
    float* alphaE = fbase + fo; fo += (size_t)nEdges * GAT_HEADS;
    float* a_max  = fbase + fo; fo += (size_t)nNodes * GAT_HEADS;
    float* a_sum  = fbase + fo; fo += (size_t)nNodes * GAT_HEADS;
    (void)ws_size; (void)n_in; (void)out_size;

    const int nx4 = nNodes * (GAT_IN_CH / 4);
    const int nw4 = (GAT_OUT_TOT * GAT_IN_CH) / 4;

    gat_init<<<(nNodes * GAT_OUT_CH + 255) / 256, 256, 0, stream>>>(out, a_max, a_sum, nNodes);
    gat_convert<<<(nx4 + 255) / 256, 256, 0, stream>>>(x, W, xh, Wh, nx4, nw4);
    gat_gemm_h<<<(nNodes + 127) / 128, 256, 0, stream>>>(xh, Wh, h16, nNodes);
    gat_alpha_node<<<(nNodes + 7) / 8, 256, 0, stream>>>(h16, att_src, att_dst, a_src, a_dst, nNodes);
    gat_edge_alpha<<<(nEdges + 255) / 256, 256, 0, stream>>>(erow, ecol, a_src, a_dst, alphaE, a_max, nEdges);
    gat_edge_exp<<<(nEdges + 255) / 256, 256, 0, stream>>>(ecol, a_max, alphaE, a_sum, nEdges);
    gat_scatter<<<(nEdges + 7) / 8, 256, 0, stream>>>(erow, ecol, h16, alphaE, a_sum, out, nEdges);
}
